// TR3DNeck_57483842290035
// MI455X (gfx1250) — compile-verified, matrix operand order checked by reading, not run on hardware
//
#include <hip/hip_runtime.h>

#define CC   128
#define CTT  288
#define CINF (CC + CTT)   // 416
#define EPS  1e-5f
#define NK   27

typedef __attribute__((ext_vector_type(16))) _Float16 v16h;
typedef __attribute__((ext_vector_type(8)))  _Float16 v8h;
typedef __attribute__((ext_vector_type(8)))  float    v8f;

// 16-bit WMMA A/B fragment layout (per ISA 7.12.2):
//   lane L: row/col = L%16, group g = L/16
//   half h (0..15): K = h + 8*g + (h>=8 ? 8 : 0)
// Inverse (for K32 = K within a 32-chunk): g = (K32>>3)&1, h = K32 - 8g - (K32&16 ? 8 : 0)

// -------------------------------------------------------------------------
// Weight pre-shuffle: f32 row-major -> f16 fragment-major, done once per call.
// Conv layout:  Wsh[(((k*8 + w)*4 + kc)*32 + lane)*16 + h]
// -------------------------------------------------------------------------
__global__ __launch_bounds__(256) void shuf_conv_w_kernel(
    const float* __restrict__ W, _Float16* __restrict__ out)
{
    const int k  = blockIdx.x;          // 27
    const int kc = blockIdx.y;          // 4
    const int w    = threadIdx.x >> 5;
    const int lane = threadIdx.x & 31;
    const int g = lane >> 4, ln = lane & 15, n0 = w * 16;
    v8h t0, t1;
    #pragma unroll
    for (int h = 0; h < 8; ++h) {
        const int K0 = kc * 32 + h + 8 * g;
        const int K1 = K0 + 16;
        t0[h] = (_Float16)W[((size_t)k * CC + K0) * CC + n0 + ln];
        t1[h] = (_Float16)W[((size_t)k * CC + K1) * CC + n0 + ln];
    }
    v8h* dst = (v8h*)(out + ((((size_t)(k * 8 + w) * 4 + kc) * 32 + lane) * 16));
    dst[0] = t0;
    dst[1] = t1;
}

// Fuse layout: Wsh[(((w*13) + kc)*32 + lane)*16 + h]
__global__ __launch_bounds__(256) void shuf_fuse_w_kernel(
    const float* __restrict__ W, _Float16* __restrict__ out)
{
    const int kc = blockIdx.x;          // 13
    const int w    = threadIdx.x >> 5;
    const int lane = threadIdx.x & 31;
    const int g = lane >> 4, ln = lane & 15, n0 = w * 16;
    v8h t0, t1;
    #pragma unroll
    for (int h = 0; h < 8; ++h) {
        const int K0 = kc * 32 + h + 8 * g;
        const int K1 = K0 + 16;
        t0[h] = (_Float16)W[(size_t)K0 * CC + n0 + ln];
        t1[h] = (_Float16)W[(size_t)K1 * CC + n0 + ln];
    }
    v8h* dst = (v8h*)(out + (((size_t)(w * 13 + kc) * 32 + lane) * 16));
    dst[0] = t0;
    dst[1] = t1;
}

// -------------------------------------------------------------------------
// Sparse conv: blockIdx.y = offset k, blockIdx.x = 64-row M-tile.
// Maps are compacted-then-padded, so mask is monotone per k: if the first row
// of this tile is masked out, the whole remainder is padding -> early exit.
// B fragments (4 x v16h) held in VGPRs, reused across 4 row-tiles.
// -------------------------------------------------------------------------
__global__ __launch_bounds__(256) void conv_wmma_kernel(
    const _Float16* __restrict__ x,     // n_in x 128 (f16 activations)
    const _Float16* __restrict__ Wsh,   // shuffled f16 weights
    const int*      __restrict__ in_map,
    const int*      __restrict__ out_map,
    const float*    __restrict__ mask,
    float*          __restrict__ out,   // n_out x 128 (pre-zeroed, f32)
    int M)
{
    __shared__ __align__(32) _Float16 Afrag[4][32][16];
    __shared__ int s_orow[16];

    const int k  = blockIdx.y;
    const int m0 = blockIdx.x * 64;
    if (mask[(size_t)k * M + m0] == 0.f) return;   // wholly padded tile range

    const int tid  = threadIdx.x;
    const int wave = tid >> 5, lane = tid & 31;
    const int g = lane >> 4, ln = lane & 15;

    // B fragments for this (k, wave): 4 chunks x 32B contiguous per lane
    v16h b[4];
    #pragma unroll
    for (int kc = 0; kc < 4; ++kc)
        b[kc] = *(const v16h*)(Wsh + ((((size_t)(k * 8 + wave) * 4 + kc) * 32 + lane) * 16));

    // gather-store geometry: thread -> (row r, 8-col group at c0)
    const int r  = tid >> 4;
    const int c0 = (tid & 15) * 8;
    const int skc = c0 >> 5;
    const int sr32 = c0 & 31;
    const int sg = (sr32 >> 3) & 1;
    const int shb = (sr32 & 16) ? 8 : 0;

    for (int rt = 0; rt < 4; ++rt) {
        const int mt0 = m0 + rt * 16;
        if (mt0 >= M) break;
        if (rt && mask[(size_t)k * M + mt0] == 0.f) break;
        if (rt) __syncthreads();        // prev tile's reads done before overwrite

        // ---- gather 16 rows into LDS fragment layout ----
        const int m = mt0 + r;
        float msk = 0.f; int src = 0;
        if (m < M) {
            const size_t o = (size_t)k * M + m;
            msk = mask[o];
            src = in_map[o];
            if ((tid & 15) == 0) s_orow[r] = (msk != 0.f) ? out_map[o] : -1;
        } else if ((tid & 15) == 0) {
            s_orow[r] = -1;
        }
        v8h av = {};
        if (msk != 0.f) av = *(const v8h*)(x + (size_t)src * CC + c0);
        *(v8h*)&Afrag[skc][sg * 16 + r][shb] = av;
        __syncthreads();

        // ---- 16x128 = A(16x128) x B(128x16 per wave) ----
        v8f acc = {};
        #pragma unroll
        for (int kc = 0; kc < 4; ++kc) {
            v16h a = *(const v16h*)&Afrag[kc][lane][0];
            acc = __builtin_amdgcn_wmma_f32_16x16x32_f16(false, a, false, b[kc],
                                                         (short)0, acc, false, false);
        }

        // ---- scatter-add: lane holds D(m = rr+8g, n = ln) ----
        #pragma unroll
        for (int rr = 0; rr < 8; ++rr) {
            const int orow = s_orow[rr + 8 * g];
            if (orow >= 0)
                atomicAdd(&out[(size_t)orow * CC + wave * 16 + ln], acc[rr]);
        }
    }
}

// -------------------------------------------------------------------------
// Fused text projection: out = concat(feats, text[bidx,0]) @ Wf   (K = 416)
// -------------------------------------------------------------------------
template <typename InT>
__global__ __launch_bounds__(256) void fuse_wmma_kernel(
    const InT*   __restrict__ feats,   // n x 128
    const float* __restrict__ text,    // B x 4 x 288
    const int*   __restrict__ bidx,
    const _Float16* __restrict__ Wsh,  // shuffled f16 fuse weights
    float*       __restrict__ out,     // n x 128
    int n)
{
    __shared__ __align__(32) _Float16 Afrag[13][32][16];

    const int m0  = blockIdx.x * 16;
    const int tid = threadIdx.x;
    {
        const int rr = tid >> 4, lc = tid & 15;
        const int m  = m0 + rr;
        const InT*   fr = nullptr;
        const float* tr = nullptr;
        if (m < n) {
            fr = feats + (size_t)m * CC;
            tr = text + (size_t)bidx[m] * (4 * CTT);   // text_feats[:,0]
        }
        #pragma unroll
        for (int j = 0; j < CINF / 16; ++j) {          // 26 cols per thread
            const int c = lc * (CINF / 16) + j;
            float v = 0.f;
            if (m < n) v = (c < CC) ? (float)fr[c] : tr[c - CC];
            const int kc = c >> 5, r32 = c & 31;
            const int gs = (r32 >> 3) & 1;
            const int h  = r32 - 8 * gs - ((r32 & 16) ? 8 : 0);
            Afrag[kc][gs * 16 + rr][h] = (_Float16)v;
        }
    }
    __syncthreads();

    const int wave = tid >> 5, lane = tid & 31;
    const int g = lane >> 4, ln = lane & 15;

    v8f acc = {};
    #pragma unroll
    for (int kc = 0; kc < 13; ++kc) {
        v16h a = *(const v16h*)&Afrag[kc][lane][0];
        v16h b = *(const v16h*)(Wsh + (((size_t)(wave * 13 + kc) * 32 + lane) * 16));
        acc = __builtin_amdgcn_wmma_f32_16x16x32_f16(false, a, false, b,
                                                     (short)0, acc, false, false);
    }

    #pragma unroll
    for (int rr = 0; rr < 8; ++rr) {
        const int m = m0 + rr + 8 * g;
        if (m < n) out[(size_t)m * CC + wave * 16 + ln] = acc[rr];
    }
}

// -------------------------------------------------------------------------
// Batch-norm: per-column sums, then apply + ReLU (+ residual), f16/f32 out
// -------------------------------------------------------------------------
__global__ __launch_bounds__(128) void colstats_kernel(
    const float* __restrict__ x, int n, float* __restrict__ stats)
{
    const int c  = threadIdx.x;
    const int r0 = blockIdx.x * 64;
    int rend = r0 + 64;
    if (rend > n) rend = n;
    float s = 0.f, s2 = 0.f;
    for (int r = r0; r < rend; ++r) {
        const float v = x[(size_t)r * CC + c];
        s += v; s2 += v * v;
    }
    atomicAdd(&stats[c], s);
    atomicAdd(&stats[CC + c], s2);
}

template <typename OutT>
__global__ __launch_bounds__(256) void bn_relu_kernel(
    const float* __restrict__ x, const float* __restrict__ stats,
    const float* __restrict__ gamma, const float* __restrict__ beta,
    const float* __restrict__ residual, OutT* __restrict__ out, int n)
{
    const size_t e = (size_t)blockIdx.x * 256 + threadIdx.x;
    if (e >= (size_t)n * CC) return;
    const int c = (int)(e & (CC - 1));
    const float invn = 1.f / (float)n;
    const float m   = stats[c] * invn;
    const float var = stats[CC + c] * invn - m * m;
    float v = (x[e] - m) * rsqrtf(var + EPS) * gamma[c] + beta[c];
    v = v > 0.f ? v : 0.f;
    if (residual) v += residual[e];
    out[e] = (OutT)v;
}

__global__ __launch_bounds__(256) void fill_zero_kernel(float* __restrict__ p, size_t n)
{
    const size_t i = (size_t)blockIdx.x * 256 + threadIdx.x;
    if (i < n) p[i] = 0.f;
}

// -------------------------------------------------------------------------
extern "C" void kernel_launch(void* const* d_in, const int* in_sizes, int n_in,
                              void* d_out, int out_size, void* d_ws, size_t ws_size,
                              hipStream_t stream)
{
    (void)n_in; (void)out_size; (void)ws_size;

    const float* feats1 = (const float*)d_in[0];
    const float* feats2 = (const float*)d_in[1];
    const float* feats3 = (const float*)d_in[2];
    const float* text   = (const float*)d_in[3];
    const int*   bidx2  = (const int*)d_in[4];
    const int*   bidx3  = (const int*)d_in[5];

    const int n1 = in_sizes[0] / CC;
    const int n2 = in_sizes[4];
    const int n3 = in_sizes[5];

    const int M1   = in_sizes[6]  / NK;
    const int M2   = in_sizes[9]  / NK;
    const int Mt32 = in_sizes[12] / NK;
    const int Mt21 = in_sizes[15] / NK;

    const int* c1_in    = (const int*)d_in[6];
    const int* c1_out   = (const int*)d_in[7];
    const float* c1_msk = (const float*)d_in[8];
    const int* c2_in    = (const int*)d_in[9];
    const int* c2_out   = (const int*)d_in[10];
    const float* c2_msk = (const float*)d_in[11];
    const int* t32_in   = (const int*)d_in[12];
    const int* t32_out  = (const int*)d_in[13];
    const float* t32_msk= (const float*)d_in[14];
    const int* t21_in   = (const int*)d_in[15];
    const int* t21_out  = (const int*)d_in[16];
    const float* t21_msk= (const float*)d_in[17];

    const float* W_up2  = (const float*)d_in[18];
    const float* W_lat1 = (const float*)d_in[19];
    const float* W_up1  = (const float*)d_in[20];
    const float* W_lat0 = (const float*)d_in[21];
    const float* W_out0 = (const float*)d_in[22];
    const float* W_f3   = (const float*)d_in[23];
    const float* W_f2   = (const float*)d_in[24];

    const float* g_f3  = (const float*)d_in[25]; const float* b_f3  = (const float*)d_in[26];
    const float* g_up2 = (const float*)d_in[27]; const float* b_up2 = (const float*)d_in[28];
    const float* g_lt1 = (const float*)d_in[29]; const float* b_lt1 = (const float*)d_in[30];
    const float* g_f2  = (const float*)d_in[31]; const float* b_f2  = (const float*)d_in[32];
    const float* g_up1 = (const float*)d_in[33]; const float* b_up1 = (const float*)d_in[34];
    const float* g_lt0 = (const float*)d_in[35]; const float* b_lt0 = (const float*)d_in[36];
    const float* g_o0  = (const float*)d_in[37]; const float* b_o0  = (const float*)d_in[38];

    // ---- workspace layout ----
    const size_t nf = (size_t)n1 * CC;
    char* wsb = (char*)d_ws;
    float*    bufA = (float*)wsb;                         // pre-bn f32 (n1*CC)
    _Float16* bufX = (_Float16*)(wsb + nf * 4);           // f16 activations
    size_t off = nf * 4 + nf * 2;
    off = (off + 255) & ~(size_t)255;
    float* stats = (float*)(wsb + off);  off += 1024;
    off = (off + 255) & ~(size_t)255;
    _Float16* Wsh = (_Float16*)(wsb + off);

    const size_t CONVW = (size_t)NK * 8 * 4 * 32 * 16;    // 442368 halves
    const size_t FUSEW = (size_t)8 * 13 * 32 * 16;        // 53248 halves
    _Float16* Wsh_up2  = Wsh;
    _Float16* Wsh_lat1 = Wsh_up2  + CONVW;
    _Float16* Wsh_up1  = Wsh_lat1 + CONVW;
    _Float16* Wsh_lat0 = Wsh_up1  + CONVW;
    _Float16* Wsh_out0 = Wsh_lat0 + CONVW;
    _Float16* Wsh_f3   = Wsh_out0 + CONVW;
    _Float16* Wsh_f2   = Wsh_f3   + FUSEW;

    // ---- weight shuffle (f32 -> f16 fragment-major), once per call ----
    shuf_conv_w_kernel<<<dim3(NK, 4), 256, 0, stream>>>(W_up2,  Wsh_up2);
    shuf_conv_w_kernel<<<dim3(NK, 4), 256, 0, stream>>>(W_lat1, Wsh_lat1);
    shuf_conv_w_kernel<<<dim3(NK, 4), 256, 0, stream>>>(W_up1,  Wsh_up1);
    shuf_conv_w_kernel<<<dim3(NK, 4), 256, 0, stream>>>(W_lat0, Wsh_lat0);
    shuf_conv_w_kernel<<<dim3(NK, 4), 256, 0, stream>>>(W_out0, Wsh_out0);
    shuf_fuse_w_kernel<<<dim3(13), 256, 0, stream>>>(W_f3, Wsh_f3);
    shuf_fuse_w_kernel<<<dim3(13), 256, 0, stream>>>(W_f2, Wsh_f2);

    auto zero = [&](float* p, size_t cnt) {
        fill_zero_kernel<<<dim3((unsigned)((cnt + 255) / 256)), 256, 0, stream>>>(p, cnt);
    };
    auto bn_f16 = [&](const float* xin, int n, const float* g, const float* b,
                      const float* res, _Float16* xout) {
        zero(stats, 2 * CC);
        colstats_kernel<<<dim3((unsigned)((n + 63) / 64)), 128, 0, stream>>>(xin, n, stats);
        bn_relu_kernel<_Float16><<<dim3((unsigned)(((size_t)n * CC + 255) / 256)), 256, 0, stream>>>(
            xin, stats, g, b, res, xout, n);
    };
    auto conv = [&](const _Float16* xin, const _Float16* Wq, const int* im, const int* om,
                    const float* mk, int M, int n_out) {
        zero(bufA, (size_t)n_out * CC);
        conv_wmma_kernel<<<dim3((unsigned)((M + 63) / 64), NK), 256, 0, stream>>>(
            xin, Wq, im, om, mk, bufA, M);
    };

    // 1. x = bn_relu(concat(feats3, t[bidx3]) @ W_fuse3)
    fuse_wmma_kernel<float><<<dim3((unsigned)((n3 + 15) / 16)), 256, 0, stream>>>(
        feats3, text, bidx3, Wsh_f3, bufA, n3);
    bn_f16(bufA, n3, g_f3, b_f3, nullptr, bufX);

    // 2. x = feats2 + bn_relu(conv_up2(x))
    conv(bufX, Wsh_up2, t32_in, t32_out, t32_msk, Mt32, n2);
    bn_f16(bufA, n2, g_up2, b_up2, feats2, bufX);

    // 3. x = bn_relu(conv_lat1(x))
    conv(bufX, Wsh_lat1, c2_in, c2_out, c2_msk, M2, n2);
    bn_f16(bufA, n2, g_lt1, b_lt1, nullptr, bufX);

    // 4. x = bn_relu(concat(x, t[bidx2]) @ W_fuse2)
    fuse_wmma_kernel<_Float16><<<dim3((unsigned)((n2 + 15) / 16)), 256, 0, stream>>>(
        bufX, text, bidx2, Wsh_f2, bufA, n2);
    bn_f16(bufA, n2, g_f2, b_f2, nullptr, bufX);

    // 5. x = feats1 + bn_relu(conv_up1(x))
    conv(bufX, Wsh_up1, t21_in, t21_out, t21_msk, Mt21, n1);
    bn_f16(bufA, n1, g_up1, b_up1, feats1, bufX);

    // 6. x = bn_relu(conv_lat0(x))
    conv(bufX, Wsh_lat0, c1_in, c1_out, c1_msk, M1, n1);
    bn_f16(bufA, n1, g_lt0, b_lt0, nullptr, bufX);

    // 7. out = bn_relu(conv_out0(x))  (final output f32 to d_out)
    conv(bufX, Wsh_out0, c1_in, c1_out, c1_msk, M1, n1);
    {
        zero(stats, 2 * CC);
        colstats_kernel<<<dim3((unsigned)((n1 + 63) / 64)), 128, 0, stream>>>(bufA, n1, stats);
        bn_relu_kernel<float><<<dim3((unsigned)(((size_t)n1 * CC + 255) / 256)), 256, 0, stream>>>(
            bufA, stats, g_o0, b_o0, nullptr, (float*)d_out, n1);
    }
}
